// CopyEditor_42279658062538
// MI455X (gfx1250) — matrix operand order repaired and checked by reading, not compiled, not run on hardware
//
#include <hip/hip_runtime.h>

// ---------------------------------------------------------------------------
// CopyEditor fused pipeline for MI455X (gfx1250, wave32, WMMA + TDM).
// B=8, NQ=1024, NC=4096, D=256, C+1=25.
// - All GEMMs on v_wmma_f32_16x16x32_bf16 with bf16 hi/lo split (3 products).
// - rel_emb GEMM stages its W tiles with the Tensor Data Mover
//   (tensor_load_to_lds + s_wait_tensorcnt) from pre-transposed bf16 planes.
// - Flash-style online softmax fuses scores -> probs -> context_vec ->
//   grouped class sums -> gate -> logaddexp in one kernel.
// ---------------------------------------------------------------------------

typedef __attribute__((ext_vector_type(16))) __bf16 v16bf;
typedef __attribute__((ext_vector_type(8)))  float  v8f;

typedef __attribute__((ext_vector_type(4))) unsigned int tdm_u32x4;
typedef __attribute__((ext_vector_type(8))) int          tdm_i32x8;
typedef __attribute__((ext_vector_type(4))) int          tdm_i32x4;

union BF16x16 { uint4 q[2]; v16bf v; unsigned short us[16]; };

__device__ __forceinline__ unsigned short f2bf(float f) {
  unsigned int u = __float_as_uint(f);
  u += 0x7fffu + ((u >> 16) & 1u);          // round-to-nearest-even
  return (unsigned short)(u >> 16);
}
__device__ __forceinline__ float bf2f(unsigned short h) {
  return __uint_as_float(((unsigned int)h) << 16);
}

__device__ __forceinline__ v8f wmma_bf16(v16bf a, v16bf b, v8f c) {
  return __builtin_amdgcn_wmma_f32_16x16x32_bf16(false, a, false, b,
                                                 (short)0, c, false, false);
}
__device__ __forceinline__ v8f wmma3(v16bf ah, v16bf al, v16bf bh, v16bf bl, v8f c) {
  c = wmma_bf16(ah, bh, c);
  c = wmma_bf16(ah, bl, c);
  c = wmma_bf16(al, bh, c);
  return c;
}

// A-frag (16x32 MxK bf16): lane holds row (l&15); elements 0-15 -> contiguous
// K = k0..k0+15 with k0 = (l<16 ? 0 : 16). Two b128 loads.
__device__ __forceinline__ v16bf frag_a_ld(const unsigned short* p) {
  BF16x16 f;
  f.q[0] = *(const uint4*)(p);
  f.q[1] = *(const uint4*)(p + 8);
  return f.v;
}
// B-frag (32x16 KxN bf16): lane holds col (l&15); elements 0-15 -> contiguous
// K = k0..k0+15 with k0 = (l<16 ? 0 : 16). Two b128 loads.
__device__ __forceinline__ v16bf frag_b_ld(const unsigned short* p) {
  BF16x16 f;
  f.q[0] = *(const uint4*)(p);
  f.q[1] = *(const uint4*)(p + 8);
  return f.v;
}

// Tensor Data Mover: 2D tile of 2-byte elements, global -> LDS.
// Descriptor packing per CDNA5 ISA 8.3/8.4: group0 {count=1, lds_addr,
// global_addr, type=2}; group1 {data_size=2B, tensor_dim0/1, tile_dim0/1,
// tensor_dim0_stride}; groups 2/3 zero (<=2D tensor).
__device__ __forceinline__ void tdm_load_2d_bf16(unsigned int lds_byte_off,
                                                 const unsigned short* gsrc,
                                                 int tensor_d0, int tensor_d1,
                                                 int tile_d0, int tile_d1,
                                                 int stride0) {
  unsigned long long ga = (unsigned long long)(size_t)gsrc;
  tdm_u32x4 g0;
  g0.x = 1u;                                              // count = 1 valid D#
  g0.y = lds_byte_off;                                    // LDS dest (bytes)
  g0.z = (unsigned int)ga;                                // global_addr[31:0]
  g0.w = (unsigned int)((ga >> 32) & 0x1FFFFFFull) | (2u << 30);  // addr hi | type=2
  tdm_i32x8 g1;
  g1[0] = (1 << 16);                                      // data_size = 2 bytes
  g1[1] = (tensor_d0 & 0xFFFF) << 16;                     // tensor_dim0[15:0]
  g1[2] = ((tensor_d0 >> 16) & 0xFFFF) | ((tensor_d1 & 0xFFFF) << 16);
  g1[3] = ((tensor_d1 >> 16) & 0xFFFF) | (tile_d0 << 16); // tile_dim0
  g1[4] = tile_d1 & 0xFFFF;                               // tile_dim1 (dim2=0)
  g1[5] = stride0;                                        // dim0 stride lo32
  g1[6] = 0;
  g1[7] = 0;
  tdm_i32x4 gz = {0, 0, 0, 0};
#if __clang_major__ >= 23
  tdm_i32x8 gz8 = {0, 0, 0, 0, 0, 0, 0, 0};
  __builtin_amdgcn_tensor_load_to_lds(g0, g1, gz, gz, gz8, 0);
#else
  __builtin_amdgcn_tensor_load_to_lds(g0, g1, gz, gz, 0);
#endif
}

__device__ __forceinline__ float log_sigmoid(float x) {
  return fminf(x, 0.f) - log1pf(expf(-fabsf(x)));
}
__device__ __forceinline__ float log_add_exp(float a, float b) {
  float m = fmaxf(a, b);
  if (m == -__builtin_inff()) return -__builtin_inff();
  return m + log1pf(expf(fminf(a, b) - m));
}

#define DD   256
#define NCC  4096
#define NQQ  1024

// ---------------------------------------------------------------------------
// Kernel 0: W_rel [512,256] fp32 -> transposed bf16 hi/lo planes Wt[n=256][k=512]
// so TDM can DMA contiguous [n][k] tiles straight into the B-frag LDS layout.
// ---------------------------------------------------------------------------
__global__ __launch_bounds__(256)
void ce_prep_w_kernel(const float* __restrict__ Wrel,
                      unsigned short* __restrict__ Wt_hi,
                      unsigned short* __restrict__ Wt_lo)
{
  int e = blockIdx.x * 256 + threadIdx.x;  // 512*256 elements
  int k = e >> 8, n = e & 255;
  float v = Wrel[e];
  unsigned short h = f2bf(v);
  Wt_hi[n * 512 + k] = h;
  Wt_lo[n * 512 + k] = f2bf(v - bf2f(h));
}

// ---------------------------------------------------------------------------
// Kernel 1: rel_emb GEMM.  Y[M,256] = [head|tail] @ W_rel + b_rel.
// Workgroup: 256 threads (8 waves, 4Mx2N), output tile 64x128.
// W tiles staged by TDM (tensor_load_to_lds); X tiles converted in-flight.
// ---------------------------------------------------------------------------
__global__ __launch_bounds__(256)
void ce_rel_emb_kernel(const float* __restrict__ head, const float* __restrict__ tail,
                       const unsigned short* __restrict__ Wt_hi,
                       const unsigned short* __restrict__ Wt_lo,
                       const float* __restrict__ brel,
                       float* __restrict__ out_f32,
                       unsigned short* __restrict__ out_hi, unsigned short* __restrict__ out_lo,
                       unsigned short* __restrict__ outT_hi, unsigned short* __restrict__ outT_lo)
{
  __shared__ __align__(16) unsigned short Xh[64][32], Xl[64][32];
  __shared__ __align__(16) unsigned short Wh[128][32], Wl[128][32];  // [n][k]

  const int tid = threadIdx.x, lane = tid & 31, wid = tid >> 5;
  const int wm = wid >> 1, wn = wid & 1;
  const int lh = lane & 15;
  const bool hih = lane >= 16;
  const int m0 = blockIdx.x * 64;
  const int nblk = blockIdx.y * 128;

  const unsigned int lds_wh = (unsigned int)(size_t)&Wh[0][0];
  const unsigned int lds_wl = (unsigned int)(size_t)&Wl[0][0];

  v8f acc[4];
#pragma unroll
  for (int s = 0; s < 4; ++s)
#pragma unroll
    for (int r = 0; r < 8; ++r) acc[s][r] = 0.f;

  for (int ks = 0; ks < 16; ++ks) {
    const int k0 = ks * 32;
    // W tile [128 n][32 k] via Tensor Data Mover (one DMA per plane, wave 0)
    if (wid == 0) {
      tdm_load_2d_bf16(lds_wh, Wt_hi + (size_t)nblk * 512 + k0,
                       512, 256, 32, 128, 512);
      tdm_load_2d_bf16(lds_wl, Wt_lo + (size_t)nblk * 512 + k0,
                       512, 256, 32, 128, 512);
    }
    // stage X tile 64x32 (fp32 -> hi/lo bf16)
#pragma unroll
    for (int i = 0; i < 8; ++i) {
      int e = tid + 256 * i;
      int row = e >> 5, kk = e & 31, gk = k0 + kk;
      float v = (gk < DD) ? head[(size_t)(m0 + row) * DD + gk]
                          : tail[(size_t)(m0 + row) * DD + (gk - DD)];
      unsigned short h = f2bf(v);
      Xh[row][kk] = h;
      Xl[row][kk] = f2bf(v - bf2f(h));
    }
    if (wid == 0) __builtin_amdgcn_s_wait_tensorcnt(0);
    __syncthreads();

    const int ka = hih ? 16 : 0;
    v16bf Ah = frag_a_ld(&Xh[wm * 16 + lh][ka]);
    v16bf Al = frag_a_ld(&Xl[wm * 16 + lh][ka]);
    const int kb = hih ? 16 : 0;
#pragma unroll
    for (int sub = 0; sub < 4; ++sub) {
      int colL = wn * 64 + sub * 16 + lh;
      v16bf Bh = frag_b_ld(&Wh[colL][kb]);
      v16bf Bl = frag_b_ld(&Wl[colL][kb]);
      acc[sub] = wmma3(Ah, Al, Bh, Bl, acc[sub]);
    }
    __syncthreads();
  }

  // epilogue: + bias, emit f32 / hi / lo / transposed
#pragma unroll
  for (int sub = 0; sub < 4; ++sub) {
    int colL = wn * 64 + sub * 16 + lh;
    int gcol = nblk + colL;
    float bias = brel[gcol];
#pragma unroll
    for (int r = 0; r < 8; ++r) {
      int grow = m0 + wm * 16 + r + (hih ? 8 : 0);
      float v = acc[sub][r] + bias;
      size_t idx = (size_t)grow * DD + gcol;
      if (out_f32) out_f32[idx] = v;
      unsigned short h = f2bf(v);
      unsigned short l = f2bf(v - bf2f(h));
      out_hi[idx] = h;
      out_lo[idx] = l;
      if (outT_hi) {  // c_emb path: [b][d][4096]
        int bb = grow >> 12, j = grow & (NCC - 1);
        size_t ti = ((size_t)(bb * DD + gcol)) * NCC + j;
        outT_hi[ti] = h;
        outT_lo[ti] = l;
      }
    }
  }
}

// ---------------------------------------------------------------------------
// Kernel 2: gen_dist = log_softmax(q_emb @ W_gen + b_gen).  One wave per row.
// ---------------------------------------------------------------------------
__global__ __launch_bounds__(32)
void ce_gen_kernel(const float* __restrict__ q_emb, const float* __restrict__ Wgen,
                   const float* __restrict__ bgen, float* __restrict__ gen_out)
{
  const int row = blockIdx.x;
  const int l = threadIdx.x;
  const int c = (l < 25) ? l : 24;
  const float* q = q_emb + (size_t)row * DD;
  float logit = 0.f;
  for (int k = 0; k < DD; ++k) logit += q[k] * Wgen[k * 25 + c];
  logit = (l < 25) ? (logit + bgen[c]) : -__builtin_inff();

  float m = logit;
#pragma unroll
  for (int off = 16; off >= 1; off >>= 1) m = fmaxf(m, __shfl_xor(m, off));
  float e = (l < 25) ? expf(logit - m) : 0.f;
  float s = e;
#pragma unroll
  for (int off = 16; off >= 1; off >>= 1) s += __shfl_xor(s, off);
  if (l < 25) gen_out[(size_t)row * 25 + l] = logit - m - logf(s);
}

// ---------------------------------------------------------------------------
// Kernel 3: fused scores -> online softmax -> context_vec (WMMA) ->
//           class sums (WMMA vs onehot) -> gate -> logaddexp output.
// Grid: B * NQ/16 workgroups; 8 waves, each strided over ctx in 32-wide chunks.
// ---------------------------------------------------------------------------
__global__ __launch_bounds__(256)
void ce_fused_attn_kernel(const unsigned short* __restrict__ q_hi,
                          const unsigned short* __restrict__ q_lo,
                          const float* __restrict__ q_f32,
                          const unsigned short* __restrict__ c_hi,
                          const unsigned short* __restrict__ c_lo,
                          const unsigned short* __restrict__ cT_hi,
                          const unsigned short* __restrict__ cT_lo,
                          const int* __restrict__ labels,
                          const float* __restrict__ mask,
                          const float* __restrict__ Wcp,
                          const float* __restrict__ bcp,
                          const float* __restrict__ gen_dist,
                          float* __restrict__ out)
{
  const int blk = blockIdx.x;
  const int b = blk >> 6;
  const int q0 = (blk & 63) * 16;          // q-row base within batch
  const size_t qrow0 = (size_t)b * NQQ + q0;

  const int tid = threadIdx.x, w = tid >> 5, l = tid & 31;
  const int lh = l & 15;
  const bool hih = l >= 16;
  const int rowoff = hih ? 8 : 0;

  __shared__ __align__(16) unsigned short Pws[8][2][16][32];  // per-wave P hi/lo
  __shared__ float ctx_sum[16][DD];
  __shared__ float cls_sum[16][32];
  __shared__ float mw_sh[8][16];
  __shared__ float mg_sh[16];

  for (int i = tid; i < 16 * DD; i += 256) ctx_sum[i >> 8][i & 255] = 0.f;
  for (int i = tid; i < 16 * 32; i += 256) cls_sum[i >> 5][i & 31] = 0.f;
  __syncthreads();

  v8f ctxacc[16], clacc[2];
#pragma unroll
  for (int t = 0; t < 16; ++t)
#pragma unroll
    for (int r = 0; r < 8; ++r) ctxacc[t][r] = 0.f;
#pragma unroll
  for (int t = 0; t < 2; ++t)
#pragma unroll
    for (int r = 0; r < 8; ++r) clacc[t][r] = 0.f;

  float m8[8];
#pragma unroll
  for (int r = 0; r < 8; ++r) m8[r] = -__builtin_inff();

  const unsigned short* qhiB = q_hi + qrow0 * DD;
  const unsigned short* qloB = q_lo + qrow0 * DD;

  for (int t = 0; t < 16; ++t) {
    const int jbase = t * 256 + w * 32;
    const int lab = labels[b * NCC + jbase + l];
    const float mk0 = mask[b * NCC + jbase + lh];
    const float mk1 = mask[b * NCC + jbase + 16 + lh];

    if (t + 1 < 16) {  // global_prefetch_b8 for next ctx chunk
      int jn = (t + 1) * 256 + w * 32;
      __builtin_prefetch(c_hi + ((size_t)(b * NCC + jn + l)) * DD, 0, 0);
    }

    // ---- scores tile 16q x 32ctx via split-bf16 WMMA over K=256 ----
    v8f S0, S1;
#pragma unroll
    for (int r = 0; r < 8; ++r) { S0[r] = 0.f; S1[r] = 0.f; }
#pragma unroll
    for (int s = 0; s < 8; ++s) {
      const int ka = s * 32 + (hih ? 16 : 0);
      v16bf Ah = frag_a_ld(qhiB + (size_t)lh * DD + ka);
      v16bf Al = frag_a_ld(qloB + (size_t)lh * DD + ka);
      const int kb = s * 32 + (hih ? 16 : 0);
      const size_t r0 = (size_t)(b * NCC + jbase + lh) * DD;
      const size_t r1 = (size_t)(b * NCC + jbase + 16 + lh) * DD;
      v16bf B0h = frag_b_ld(c_hi + r0 + kb);
      v16bf B0l = frag_b_ld(c_lo + r0 + kb);
      v16bf B1h = frag_b_ld(c_hi + r1 + kb);
      v16bf B1l = frag_b_ld(c_lo + r1 + kb);
      S0 = wmma3(Ah, Al, B0h, B0l, S0);
      S1 = wmma3(Ah, Al, B1h, B1l, S1);
    }
#pragma unroll
    for (int r = 0; r < 8; ++r) { S0[r] *= mk0; S1[r] *= mk1; }

    // ---- online softmax: per-row max across the 16 lanes of each half ----
    float sc[8];
#pragma unroll
    for (int r = 0; r < 8; ++r) {
      float v = fmaxf(S0[r], S1[r]);
#pragma unroll
      for (int off = 8; off >= 1; off >>= 1) v = fmaxf(v, __shfl_xor(v, off));
      float mn = fmaxf(m8[r], v);
      sc[r] = __expf(m8[r] - mn);
      m8[r] = mn;
    }
#pragma unroll
    for (int t16 = 0; t16 < 16; ++t16)
#pragma unroll
      for (int r = 0; r < 8; ++r) ctxacc[t16][r] *= sc[r];
#pragma unroll
    for (int ct = 0; ct < 2; ++ct)
#pragma unroll
      for (int r = 0; r < 8; ++r) clacc[ct][r] *= sc[r];

    // ---- P = exp(S - m); stash hi/lo bf16 in per-wave LDS (C-frag layout) --
#pragma unroll
    for (int r = 0; r < 8; ++r) {
      float p0 = __expf(S0[r] - m8[r]);
      float p1 = __expf(S1[r] - m8[r]);
      unsigned short h0 = f2bf(p0), h1 = f2bf(p1);
      Pws[w][0][r + rowoff][lh]      = h0;
      Pws[w][0][r + rowoff][lh + 16] = h1;
      Pws[w][1][r + rowoff][lh]      = f2bf(p0 - bf2f(h0));
      Pws[w][1][r + rowoff][lh + 16] = f2bf(p1 - bf2f(h1));
    }
    // same-wave DS ops are in-order; compiler inserts s_wait_dscnt.
    const int kpa = hih ? 16 : 0;
    v16bf Pah = frag_a_ld(&Pws[w][0][lh][kpa]);
    v16bf Pal = frag_a_ld(&Pws[w][1][lh][kpa]);

    // ---- context_vec accumulation: P[16x32] @ c_emb[32x256] ----
    const int kjb = jbase + (hih ? 16 : 0);
#pragma unroll
    for (int t16 = 0; t16 < 16; ++t16) {
      const int d = t16 * 16 + lh;
      const size_t rT = (size_t)(b * DD + d) * NCC + kjb;
      v16bf Ch = frag_b_ld(cT_hi + rT);
      v16bf Cl = frag_b_ld(cT_lo + rT);
      ctxacc[t16] = wmma_bf16(Pah, Ch, ctxacc[t16]);
      ctxacc[t16] = wmma_bf16(Pal, Ch, ctxacc[t16]);
      ctxacc[t16] = wmma_bf16(Pah, Cl, ctxacc[t16]);
    }

    // ---- grouped class sums: P[16x32] @ onehot[32x32] ----
#pragma unroll
    for (int ct = 0; ct < 2; ++ct) {
      const int cls = ct * 16 + lh;
      BF16x16 oh;
      const int jo = hih ? 16 : 0;
#pragma unroll
      for (int i = 0; i < 16; ++i) {
        int lv = __shfl(lab, jo + i);
        oh.us[i] = (lv == cls) ? (unsigned short)0x3F80 : (unsigned short)0;
      }
      clacc[ct] = wmma_bf16(Pah, oh.v, clacc[ct]);
      clacc[ct] = wmma_bf16(Pal, oh.v, clacc[ct]);
    }
  }

  // ---- cross-wave flash combine ----
  if (lh == 0) {
#pragma unroll
    for (int r = 0; r < 8; ++r) mw_sh[w][r + rowoff] = m8[r];
  }
  __syncthreads();
  if (tid < 16) {
    float mm = -__builtin_inff();
    for (int ww = 0; ww < 8; ++ww) mm = fmaxf(mm, mw_sh[ww][tid]);
    mg_sh[tid] = mm;
  }
  __syncthreads();

  float sc[8];
#pragma unroll
  for (int r = 0; r < 8; ++r) sc[r] = __expf(m8[r] - mg_sh[r + rowoff]);
#pragma unroll
  for (int t16 = 0; t16 < 16; ++t16)
#pragma unroll
    for (int r = 0; r < 8; ++r)
      atomicAdd(&ctx_sum[r + rowoff][t16 * 16 + lh], ctxacc[t16][r] * sc[r]);
#pragma unroll
  for (int ct = 0; ct < 2; ++ct)
#pragma unroll
    for (int r = 0; r < 8; ++r)
      atomicAdd(&cls_sum[r + rowoff][ct * 16 + lh], clacc[ct][r] * sc[r]);
  __syncthreads();

  // ---- epilogue: gate + logaddexp, wave w handles rows w and w+8 ----
#pragma unroll
  for (int pass = 0; pass < 2; ++pass) {
    const int row = w + pass * 8;
    const size_t qg = qrow0 + row;

    float part = (l < 25) ? cls_sum[row][l] : 0.f;
#pragma unroll
    for (int off = 16; off >= 1; off >>= 1) part += __shfl_xor(part, off);
    const float lsum = part;
    const float logl = logf(lsum);

    float g = 0.f;
    for (int d = l; d < DD; d += 32)
      g += q_f32[qg * DD + d] * Wcp[d] + (ctx_sum[row][d] / lsum) * Wcp[DD + d];
#pragma unroll
    for (int off = 16; off >= 1; off >>= 1) g += __shfl_xor(g, off);
    const float logit = g + bcp[0];
    const float lcp = log_sigmoid(logit);
    const float lgen = log_sigmoid(-logit);

    if (l < 25) {
      float cd = logf(cls_sum[row][l]) - logl;   // log(0) = -inf for empty class
      float gd = gen_dist[qg * 25 + l];
      out[qg * 25 + l] = log_add_exp(lcp + cd, lgen + gd);
    }
  }
}

// ---------------------------------------------------------------------------
// Host launcher
// ---------------------------------------------------------------------------
extern "C" void kernel_launch(void* const* d_in, const int* in_sizes, int n_in,
                              void* d_out, int out_size, void* d_ws, size_t ws_size,
                              hipStream_t stream) {
  const float* q_head = (const float*)d_in[0];
  const float* q_tail = (const float*)d_in[1];
  const float* c_head = (const float*)d_in[2];
  const float* c_tail = (const float*)d_in[3];
  const int*   labels = (const int*)d_in[4];
  const float* mask   = (const float*)d_in[5];
  const float* W_rel  = (const float*)d_in[6];
  const float* b_rel  = (const float*)d_in[7];
  const float* W_gen  = (const float*)d_in[8];
  const float* b_gen  = (const float*)d_in[9];
  const float* W_cp   = (const float*)d_in[10];
  const float* b_cp   = (const float*)d_in[11];
  float* out = (float*)d_out;

  const size_t Mq = 8 * 1024, Mc = 8 * 4096;
  char* ws = (char*)d_ws;
  float* q_emb            = (float*)ws;           ws += Mq * DD * sizeof(float);
  unsigned short* q_hi    = (unsigned short*)ws;  ws += Mq * DD * 2;
  unsigned short* q_lo    = (unsigned short*)ws;  ws += Mq * DD * 2;
  unsigned short* c_hi    = (unsigned short*)ws;  ws += Mc * DD * 2;
  unsigned short* c_lo    = (unsigned short*)ws;  ws += Mc * DD * 2;
  unsigned short* cT_hi   = (unsigned short*)ws;  ws += Mc * DD * 2;
  unsigned short* cT_lo   = (unsigned short*)ws;  ws += Mc * DD * 2;
  float* gen_dist         = (float*)ws;           ws += Mq * 25 * sizeof(float);
  unsigned short* Wt_hi   = (unsigned short*)ws;  ws += 512 * 256 * 2;
  unsigned short* Wt_lo   = (unsigned short*)ws;  ws += 512 * 256 * 2;

  ce_prep_w_kernel<<<dim3(512), 256, 0, stream>>>(W_rel, Wt_hi, Wt_lo);
  ce_rel_emb_kernel<<<dim3(Mq / 64, 2), 256, 0, stream>>>(
      q_head, q_tail, Wt_hi, Wt_lo, b_rel, q_emb, q_hi, q_lo, nullptr, nullptr);
  ce_rel_emb_kernel<<<dim3(Mc / 64, 2), 256, 0, stream>>>(
      c_head, c_tail, Wt_hi, Wt_lo, b_rel, nullptr, c_hi, c_lo, cT_hi, cT_lo);
  ce_gen_kernel<<<dim3(Mq), 32, 0, stream>>>(q_emb, W_gen, b_gen, gen_dist);
  ce_fused_attn_kernel<<<dim3(8 * 64), 256, 0, stream>>>(
      q_hi, q_lo, q_emb, c_hi, c_lo, cT_hi, cT_lo, labels, mask,
      W_cp, b_cp, gen_dist, out);
}
